// Model_41592463294537
// MI455X (gfx1250) — matrix operand order repaired and checked by reading, not compiled, not run on hardware
//
#include <hip/hip_runtime.h>
#include <hip/hip_bf16.h>

// ---------------------------------------------------------------- constants
#define Hdim   1024
#define Bdim   32
#define Tdim   128
#define Vdim   32000
#define MOG    5

typedef __bf16 bf16;
typedef __attribute__((ext_vector_type(16))) bf16  v16bf;
typedef __attribute__((ext_vector_type(8)))  bf16  v8bf;
typedef __attribute__((ext_vector_type(8)))  float v8f;

// ---------------------------------------------------------------- helpers
// Fast sigmoid/tanh: v_exp_f32 + v_rcp_f32 (no IEEE division expansion).
__device__ __forceinline__ float sigmoidf_(float x) {
    return __builtin_amdgcn_rcpf(1.0f + __expf(-x));
}
__device__ __forceinline__ float tanhf_(float x) {
    // 2*sigmoid(2x)-1; saturates cleanly to +/-1 for large |x|.
    return 2.0f * __builtin_amdgcn_rcpf(1.0f + __expf(-2.0f * x)) - 1.0f;
}

// A-matrix tile (16x32 bf16, rows m0..m0+15, K slice k0..k0+31) from a
// row-major bf16 activation matrix [rows x lda].
// Layout (16-bit A 16x32): lanes 0-15 hold K 0..15 of row m0+lane,
// lanes 16-31 hold K 16..31 of row m0+lane-16.
__device__ __forceinline__ v16bf load_A_tile(const bf16* __restrict__ A, int lda,
                                             int m0, int k0, int lane) {
    const int half = lane >> 4;            // 0 or 1
    const int row  = m0 + (lane & 15);
    const bf16* base = A + (size_t)row * lda + k0 + half * 16;
    v8bf lo = *(const v8bf*)(base);        // K +0..7   (or +16..23)
    v8bf hi = *(const v8bf*)(base + 8);    // K +8..15  (or +24..31)
    v16bf a;
#pragma unroll
    for (int i = 0; i < 8; ++i) { a[i] = lo[i]; a[8 + i] = hi[i]; }
    return a;
}

// B-matrix tile (32x16 bf16, K slice k0..k0+31, cols n0..n0+15) from a
// row-major [N x K] weight (PyTorch convention, y = x @ W^T).
// Same contiguous k order as load_A_tile: lanes 0-15 hold K 0..15 of column n0+lane, lanes 16-31 hold
// K 16..31 of column n0+lane-16  ->  one 32-byte contiguous load per lane.
__device__ __forceinline__ v16bf load_B_tile(const bf16* __restrict__ W, int ldw,
                                             int n0, int k0, int lane) {
    const int col  = lane & 15;
    const int koff = (lane >> 4) * 16;
    return *(const v16bf*)(W + (size_t)(n0 + col) * ldw + k0 + koff);
}

__device__ __forceinline__ v8f wmma_bf16(v16bf a, v16bf b, v8f c) {
    return __builtin_amdgcn_wmma_f32_16x16x32_bf16(false, a, false, b,
                                                   (short)0, c, false, false);
}

// D/C tile mapping: VGPR r, lane l -> row m0 + r + (l>=16 ? 8 : 0), col n0 + (l&15)

// ---------------------------------------------------------------- one-time prep
__global__ void k_f32_to_bf16(const float* __restrict__ src, bf16* __restrict__ dst,
                              long n) {
    long i = (long)blockIdx.x * blockDim.x + threadIdx.x;
    long stride = (long)gridDim.x * blockDim.x;
    for (; i < n; i += stride) dst[i] = (bf16)src[i];
}

__global__ void k_bias_combine(const float* __restrict__ a, const float* __restrict__ b,
                               float* __restrict__ o, int n) {
    int i = blockIdx.x * blockDim.x + threadIdx.x;
    if (i < n) o[i] = a[i] + b[i];
}

__global__ void k_zero_f32(float* __restrict__ p, long n) {
    long i = (long)blockIdx.x * blockDim.x + threadIdx.x;
    long stride = (long)gridDim.x * blockDim.x;
    for (; i < n; i += stride) p[i] = 0.0f;
}

// ---------------------------------------------------------------- per-timestep
// xw1 <- emb[seq[:,t]] (f32 + bf16 mirrors); hw1 <- h1 (f32 + bf16 mirrors)
__global__ void k_embed_init(const int* __restrict__ seq, const float* __restrict__ emb,
                             int t,
                             float* __restrict__ xw_f, bf16* __restrict__ xw_bf,
                             const float* __restrict__ h_f,
                             float* __restrict__ hw_f, bf16* __restrict__ hw_bf) {
    int i = blockIdx.x * blockDim.x + threadIdx.x;       // 0 .. 32*1024-1
    if (i >= Bdim * Hdim) return;
    int b = i >> 10, col = i & (Hdim - 1);
    int tok = seq[b * Tdim + t];
    float v = emb[(size_t)tok * Hdim + col];
    xw_f[i] = v; xw_bf[i] = (bf16)v;
    float hv = h_f[i];
    hw_f[i] = hv; hw_bf[i] = (bf16)hv;
}

// One mogrifier step:  tgt *= 2*sigmoid(src @ W^T + bias)   (in place, f32+bf16)
// 128 wave-jobs: 2 M-tiles x 64 N-tiles. Launch <<<16, 256>>>.
__global__ __launch_bounds__(256) void k_mogrify(const bf16* __restrict__ src_bf,
                                                 float* __restrict__ tgt_f,
                                                 bf16* __restrict__ tgt_bf,
                                                 const bf16* __restrict__ W,
                                                 const float* __restrict__ bias) {
    int wave = (blockIdx.x * blockDim.x + threadIdx.x) >> 5;   // 0..127
    int lane = threadIdx.x & 31;
    int m0 = (wave & 1) * 16;
    int n0 = (wave >> 1) * 16;
    v8f acc = {};
#pragma unroll 4
    for (int k0 = 0; k0 < Hdim; k0 += 32) {
        v16bf a = load_A_tile(src_bf, Hdim, m0, k0, lane);
        v16bf b = load_B_tile(W, Hdim, n0, k0, lane);
        acc = wmma_bf16(a, b, acc);
    }
    int col   = n0 + (lane & 15);
    int rbase = m0 + ((lane >> 4) ? 8 : 0);
    float bsv = bias[col];
#pragma unroll
    for (int r = 0; r < 8; ++r) {
        int idx = (rbase + r) * Hdim + col;
        float gate = 2.0f * sigmoidf_(acc[r] + bsv);
        float nv = gate * tgt_f[idx];
        tgt_f[idx]  = nv;
        tgt_bf[idx] = (bf16)nv;
    }
}

// Fused LSTM: gates = xw@Wih^T + hw@Whh^T + bias ; pointwise i,f,g,o -> c,h.
// Each wave owns (m-tile, h-column tile) and computes the 4 gate tiles for
// that column slice so i/f/g/o are lane-aligned. 128 wave-jobs, <<<16,256>>>.
__global__ __launch_bounds__(256) void k_lstm(const bf16* __restrict__ xw_bf,
                                              const bf16* __restrict__ hw_bf,
                                              const bf16* __restrict__ Wih,   // [4H][H]
                                              const bf16* __restrict__ Whh,   // [4H][H]
                                              const float* __restrict__ bias, // [4H]
                                              float* __restrict__ c_f,        // in/out
                                              float* __restrict__ h_f,        // out
                                              float* __restrict__ nx_f,       // nullable: next cell x working copy
                                              bf16*  __restrict__ nx_bf,
                                              const float* __restrict__ oh_f, // nullable: other cell's persistent h
                                              float* __restrict__ nh_f,       //   -> its working copy
                                              bf16*  __restrict__ nh_bf,
                                              float* __restrict__ hist_f,     // nullable: hidden_states [B][T][H]
                                              bf16*  __restrict__ abuf,       // nullable: GEMM A rows (b*T+t)
                                              int t) {
    int wave = (blockIdx.x * blockDim.x + threadIdx.x) >> 5;   // 0..127
    int lane = threadIdx.x & 31;
    int m0  = (wave & 1) * 16;
    int n0h = (wave >> 1) * 16;            // h-column tile base, 0..1008
    v8f acc[4] = {v8f{}, v8f{}, v8f{}, v8f{}};
#pragma unroll 2
    for (int k0 = 0; k0 < Hdim; k0 += 32) {
        v16bf a = load_A_tile(xw_bf, Hdim, m0, k0, lane);
#pragma unroll
        for (int g = 0; g < 4; ++g) {
            v16bf b = load_B_tile(Wih, Hdim, g * Hdim + n0h, k0, lane);
            acc[g] = wmma_bf16(a, b, acc[g]);
        }
    }
#pragma unroll 2
    for (int k0 = 0; k0 < Hdim; k0 += 32) {
        v16bf a = load_A_tile(hw_bf, Hdim, m0, k0, lane);
#pragma unroll
        for (int g = 0; g < 4; ++g) {
            v16bf b = load_B_tile(Whh, Hdim, g * Hdim + n0h, k0, lane);
            acc[g] = wmma_bf16(a, b, acc[g]);
        }
    }
    int col   = n0h + (lane & 15);
    int rbase = m0 + ((lane >> 4) ? 8 : 0);
    float bi = bias[col], bf_ = bias[Hdim + col], bg = bias[2 * Hdim + col],
          bo = bias[3 * Hdim + col];
#pragma unroll
    for (int r = 0; r < 8; ++r) {
        int row = rbase + r;               // == batch index b
        int idx = row * Hdim + col;
        float ig = sigmoidf_(acc[0][r] + bi);
        float fg = sigmoidf_(acc[1][r] + bf_);
        float gg = tanhf_(acc[2][r] + bg);
        float og = sigmoidf_(acc[3][r] + bo);
        float c  = fg * c_f[idx] + ig * gg;
        c_f[idx] = c;
        float h  = og * tanhf_(c);
        h_f[idx] = h;
        if (nx_f)   { nx_f[idx] = h; nx_bf[idx] = (bf16)h; }
        if (oh_f)   { float o = oh_f[idx]; nh_f[idx] = o; nh_bf[idx] = (bf16)o; }
        size_t hidx = ((size_t)row * Tdim + t) * Hdim + col;
        if (hist_f) hist_f[hidx] = h;
        if (abuf)   abuf[hidx]   = (bf16)h;
    }
}

// ---------------------------------------------------------------- final GEMM
// out[B*T, V] = A[B*T, H] @ emb^T + fc_b.
// Block = one N-group (64 emb rows) x 8 M-tiles (one per wave). The 64x1024
// bf16 B panel is staged through LDS in four 32 KB K-slabs shared by all 8
// waves (8x cut in B traffic); A streams from L2 per wave.
// Grid: 500 N-groups * 32 M-blocks = 16000 blocks of 256 threads.
#define PROJ_KSLAB 256
__global__ __launch_bounds__(256) void k_proj(const bf16* __restrict__ A,
                                              const bf16* __restrict__ Wemb, // [V][H]
                                              const float* __restrict__ fc_b,
                                              float* __restrict__ out) {
    __shared__ bf16 sB[64 * PROJ_KSLAB];   // 32 KB

    const int lane  = threadIdx.x & 31;
    const int waveI = threadIdx.x >> 5;    // 0..7
    const int ng = blockIdx.x >> 5;        // 0..499
    const int mb = blockIdx.x & 31;        // 0..31
    const int n0 = ng * 64;
    const int m0 = (mb * 8 + waveI) * 16;  // this wave's M-tile

    // cooperative-fill mapping: thread -> (column, 128B chunk)
    const int fcol  = threadIdx.x >> 2;          // 0..63
    const int fpart = threadIdx.x & 3;           // 0..3  (64 bf16 each)

    v8f acc[4] = {v8f{}, v8f{}, v8f{}, v8f{}};

    for (int s = 0; s < Hdim / PROJ_KSLAB; ++s) {
        const int kb = s * PROJ_KSLAB;
        if (s) __syncthreads();            // previous slab fully consumed
        // stage B[n0..n0+63][kb..kb+255] into LDS (each thread: 128 bytes)
        {
            const uint4* gsrc =
                (const uint4*)(Wemb + (size_t)(n0 + fcol) * Hdim + kb + fpart * 64);
            uint4* ldst = (uint4*)(&sB[fcol * PROJ_KSLAB + fpart * 64]);
#pragma unroll
            for (int q = 0; q < 8; ++q) ldst[q] = gsrc[q];
        }
        if (s + 1 < Hdim / PROJ_KSLAB) {   // pull next slab toward L0/L2
            __builtin_prefetch(Wemb + (size_t)(n0 + fcol) * Hdim + kb + PROJ_KSLAB +
                                   fpart * 64, 0, 1);
        }
        __syncthreads();

        const int colB = lane & 15;
        const int koff = (lane >> 4) * 16;
#pragma unroll 2
        for (int kl = 0; kl < PROJ_KSLAB; kl += 32) {
            v16bf a = load_A_tile(A, Hdim, m0, kb + kl, lane);
#pragma unroll
            for (int j = 0; j < 4; ++j) {
                v16bf b = *(const v16bf*)(&sB[(j * 16 + colB) * PROJ_KSLAB + kl + koff]);
                acc[j] = wmma_bf16(a, b, acc[j]);
            }
        }
    }

    const int rbase = m0 + ((lane >> 4) ? 8 : 0);
#pragma unroll
    for (int j = 0; j < 4; ++j) {
        int col = n0 + j * 16 + (lane & 15);
        float bv = fc_b[col];
#pragma unroll
        for (int r = 0; r < 8; ++r) {
            out[(size_t)(rbase + r) * Vdim + col] = acc[j][r] + bv;
        }
    }
}

// ---------------------------------------------------------------- launcher
extern "C" void kernel_launch(void* const* d_in, const int* in_sizes, int n_in,
                              void* d_out, int out_size, void* d_ws, size_t ws_size,
                              hipStream_t stream) {
    (void)in_sizes; (void)n_in; (void)out_size; (void)ws_size;
    const int*   seq      = (const int*)  d_in[0];
    const float* emb      = (const float*)d_in[2];
    const float* fc_b     = (const float*)d_in[3];
    const float* c1_mog_w = (const float*)d_in[4];
    const float* c1_mog_b = (const float*)d_in[5];
    const float* c1_wih   = (const float*)d_in[6];
    const float* c1_whh   = (const float*)d_in[7];
    const float* c1_bih   = (const float*)d_in[8];
    const float* c1_bhh   = (const float*)d_in[9];
    const float* c2_mog_w = (const float*)d_in[10];
    const float* c2_mog_b = (const float*)d_in[11];
    const float* c2_wih   = (const float*)d_in[12];
    const float* c2_whh   = (const float*)d_in[13];
    const float* c2_bih   = (const float*)d_in[14];
    const float* c2_bhh   = (const float*)d_in[15];

    float* out_logits = (float*)d_out;                               // [B][T][V]
    float* out_hidden = out_logits + (size_t)Bdim * Tdim * Vdim;     // [B][T][H]

    // ---- workspace carve-up (256B aligned) ----
    char* ws = (char*)d_ws;
    auto take = [&](size_t bytes) {
        char* p = ws;
        ws += (bytes + 255) & ~(size_t)255;
        return p;
    };
    bf16* emb_bf  = (bf16*)take((size_t)Vdim * Hdim * 2);
    bf16* mog1_bf = (bf16*)take((size_t)MOG * Hdim * Hdim * 2);
    bf16* mog2_bf = (bf16*)take((size_t)MOG * Hdim * Hdim * 2);
    bf16* wih1_bf = (bf16*)take((size_t)4 * Hdim * Hdim * 2);
    bf16* whh1_bf = (bf16*)take((size_t)4 * Hdim * Hdim * 2);
    bf16* wih2_bf = (bf16*)take((size_t)4 * Hdim * Hdim * 2);
    bf16* whh2_bf = (bf16*)take((size_t)4 * Hdim * Hdim * 2);
    float* bias1  = (float*)take((size_t)4 * Hdim * 4);
    float* bias2  = (float*)take((size_t)4 * Hdim * 4);
    bf16* Abuf    = (bf16*)take((size_t)Bdim * Tdim * Hdim * 2);     // h2 history rows b*T+t
    float* xw1_f  = (float*)take((size_t)Bdim * Hdim * 4);
    bf16*  xw1_bf = (bf16*) take((size_t)Bdim * Hdim * 2);
    float* hw1_f  = (float*)take((size_t)Bdim * Hdim * 4);
    bf16*  hw1_bf = (bf16*) take((size_t)Bdim * Hdim * 2);
    float* xw2_f  = (float*)take((size_t)Bdim * Hdim * 4);
    bf16*  xw2_bf = (bf16*) take((size_t)Bdim * Hdim * 2);
    float* hw2_f  = (float*)take((size_t)Bdim * Hdim * 4);
    bf16*  hw2_bf = (bf16*) take((size_t)Bdim * Hdim * 2);
    float* h1_f   = (float*)take((size_t)Bdim * Hdim * 4);
    float* c1_f   = (float*)take((size_t)Bdim * Hdim * 4);
    float* h2_f   = (float*)take((size_t)Bdim * Hdim * 4);
    float* c2_f   = (float*)take((size_t)Bdim * Hdim * 4);

    // ---- one-time prep ----
    k_zero_f32<<<128, 256, 0, stream>>>(h1_f, (long)4 * Bdim * Hdim); // h1,c1,h2,c2 contiguous
    k_f32_to_bf16<<<2048, 256, 0, stream>>>(emb,      emb_bf,  (long)Vdim * Hdim);
    k_f32_to_bf16<<<1024, 256, 0, stream>>>(c1_mog_w, mog1_bf, (long)MOG * Hdim * Hdim);
    k_f32_to_bf16<<<1024, 256, 0, stream>>>(c2_mog_w, mog2_bf, (long)MOG * Hdim * Hdim);
    k_f32_to_bf16<<<1024, 256, 0, stream>>>(c1_wih,   wih1_bf, (long)4 * Hdim * Hdim);
    k_f32_to_bf16<<<1024, 256, 0, stream>>>(c1_whh,   whh1_bf, (long)4 * Hdim * Hdim);
    k_f32_to_bf16<<<1024, 256, 0, stream>>>(c2_wih,   wih2_bf, (long)4 * Hdim * Hdim);
    k_f32_to_bf16<<<1024, 256, 0, stream>>>(c2_whh,   whh2_bf, (long)4 * Hdim * Hdim);
    k_bias_combine<<<16, 256, 0, stream>>>(c1_bih, c1_bhh, bias1, 4 * Hdim);
    k_bias_combine<<<16, 256, 0, stream>>>(c2_bih, c2_bhh, bias2, 4 * Hdim);

    // ---- sequential recurrence ----
    for (int t = 0; t < Tdim; ++t) {
        k_embed_init<<<(Bdim * Hdim + 255) / 256, 256, 0, stream>>>(
            seq, emb, t, xw1_f, xw1_bf, h1_f, hw1_f, hw1_bf);

        // cell 1 mogrify (even step gates x from h, odd gates h from x)
        for (int i = 0; i < MOG; ++i) {
            if ((i & 1) == 0)
                k_mogrify<<<16, 256, 0, stream>>>(hw1_bf, xw1_f, xw1_bf,
                                                  mog1_bf + (size_t)i * Hdim * Hdim,
                                                  c1_mog_b + i * Hdim);
            else
                k_mogrify<<<16, 256, 0, stream>>>(xw1_bf, hw1_f, hw1_bf,
                                                  mog1_bf + (size_t)i * Hdim * Hdim,
                                                  c1_mog_b + i * Hdim);
        }
        // cell 1 LSTM; also seeds cell-2 working buffers (xw2 <- h1, hw2 <- h2)
        k_lstm<<<16, 256, 0, stream>>>(xw1_bf, hw1_bf, wih1_bf, whh1_bf, bias1,
                                       c1_f, h1_f,
                                       xw2_f, xw2_bf,
                                       h2_f, hw2_f, hw2_bf,
                                       nullptr, nullptr, t);

        // cell 2 mogrify
        for (int i = 0; i < MOG; ++i) {
            if ((i & 1) == 0)
                k_mogrify<<<16, 256, 0, stream>>>(hw2_bf, xw2_f, xw2_bf,
                                                  mog2_bf + (size_t)i * Hdim * Hdim,
                                                  c2_mog_b + i * Hdim);
            else
                k_mogrify<<<16, 256, 0, stream>>>(xw2_bf, hw2_f, hw2_bf,
                                                  mog2_bf + (size_t)i * Hdim * Hdim,
                                                  c2_mog_b + i * Hdim);
        }
        // cell 2 LSTM; records h2 into hidden_states output and the GEMM A buffer
        k_lstm<<<16, 256, 0, stream>>>(xw2_bf, hw2_bf, wih2_bf, whh2_bf, bias2,
                                       c2_f, h2_f,
                                       nullptr, nullptr,
                                       nullptr, nullptr, nullptr,
                                       out_hidden, Abuf, t);
    }

    // ---- hoisted tied output projection: [4096,1024] x [1024,32000] ----
    k_proj<<<16000, 256, 0, stream>>>(Abuf, emb_bf, fc_b, out_logits);
}
